// SwinTransformerBlock_43276090474686
// MI455X (gfx1250) — compile-verified, matrix operand order checked by reading, not run on hardware
//
#include <hip/hip_runtime.h>
#include <math.h>

// ---------------------------------------------------------------------------
// Types for CDNA5 WMMA (wave32, 16x16x32 bf16 -> f32)
// ---------------------------------------------------------------------------
typedef __bf16 bf16;
typedef __attribute__((ext_vector_type(16))) __bf16 v16bf;
typedef __attribute__((ext_vector_type(8)))  float  v8f;

#define DIMC    512
#define HEADS_  16
#define HD      32
#define NP      80      // window tokens padded 65 -> 80 (multiple of 16)
#define NVALID  65
#define SHIFT_  4
#define HIDDEN_ 2048

__device__ __forceinline__ v8f v8f_zero() {
  v8f z;
  for (int i = 0; i < 8; ++i) z[i] = 0.0f;
  return z;
}

// ---------------------------------------------------------------------------
// Generic tiled bf16 WMMA GEMM:  C[M,N] = A[M,K] * Bt[N,K]^T   (Bt row n = col n)
// Block: 256 threads = 8 waves, tile 128x128, K-step 32.
// A/B tiles staged with CDNA5 async global->LDS DMA (ASYNCcnt), no VGPR bounce.
// Wave grid 4(M) x 2(N): each wave owns 32x64 = 2x4 wmma accumulators.
// ---------------------------------------------------------------------------
template <typename Epi>
__global__ __launch_bounds__(256) void gemm_bf16_wmma(
    const bf16* __restrict__ A, const bf16* __restrict__ Bt,
    int M, int N, int K, Epi epi)
{
  __shared__ __attribute__((aligned(64))) bf16 As[128 * 32];
  __shared__ __attribute__((aligned(64))) bf16 Bs[128 * 32];

  const int tid  = threadIdx.x;
  const int lane = tid & 31;
  const int wid  = tid >> 5;
  const int wm   = wid >> 1;          // 0..3
  const int wn   = wid & 1;           // 0..1
  const int row0 = blockIdx.y * 128;
  const int col0 = blockIdx.x * 128;

  v8f acc[2][4];
  for (int i = 0; i < 2; ++i)
    for (int j = 0; j < 4; ++j) acc[i][j] = v8f_zero();

  const int lr   = lane & 15;
  const int koff = (lane >> 4) * 16;
  const int ldr  = tid >> 1;          // 0..127 (tile row to stage)
  const int ldh  = (tid & 1) * 16;    // 16-element half of the 32-wide K slab

  // LDS byte addresses for this thread's staging slot (flat low 32 bits = LDS addr)
  const unsigned lds_a = (unsigned)(size_t)(As + ldr * 32 + ldh);
  const unsigned lds_b = (unsigned)(size_t)(Bs + ldr * 32 + ldh);

  for (int k0 = 0; k0 < K; k0 += 32) {
    // async DMA: 32 bytes per thread per tile, INST_OFFSET applies to both sides
    unsigned long long ga = (unsigned long long)(size_t)
        (A + (size_t)(row0 + ldr) * K + k0 + ldh);
    unsigned long long gb = (unsigned long long)(size_t)
        (Bt + (size_t)(col0 + ldr) * K + k0 + ldh);
    asm volatile(
        "global_load_async_to_lds_b128 %0, %2, off\n\t"
        "global_load_async_to_lds_b128 %0, %2, off offset:16\n\t"
        "global_load_async_to_lds_b128 %1, %3, off\n\t"
        "global_load_async_to_lds_b128 %1, %3, off offset:16"
        :: "v"(lds_a), "v"(lds_b), "v"(ga), "v"(gb) : "memory");
    if (k0 + 32 < K) {   // pull next K-slab toward the caches (global_prefetch_b8)
      __builtin_prefetch(A  + (size_t)(row0 + ldr) * K + k0 + 32 + ldh, 0, 3);
      __builtin_prefetch(Bt + (size_t)(col0 + ldr) * K + k0 + 32 + ldh, 0, 3);
    }
    asm volatile("s_wait_asynccnt 0x0" ::: "memory");
    __syncthreads();

    v16bf af[2], bfv[4];
    for (int i = 0; i < 2; ++i)
      af[i]  = *(const v16bf*)(As + (wm * 32 + i * 16 + lr) * 32 + koff);
    for (int j = 0; j < 4; ++j)
      bfv[j] = *(const v16bf*)(Bs + (wn * 64 + j * 16 + lr) * 32 + koff);

    for (int i = 0; i < 2; ++i)
      for (int j = 0; j < 4; ++j)
        acc[i][j] = __builtin_amdgcn_wmma_f32_16x16x32_bf16(
            false, af[i], false, bfv[j], (short)0, acc[i][j], false, false);
    __syncthreads();
  }

  // C-fragment layout: vgpr r -> M = r (lanes 0-15) / r+8 (lanes 16-31), N = lane&15
  const int mh = (lane >> 4) * 8;
  for (int i = 0; i < 2; ++i)
    for (int j = 0; j < 4; ++j)
      for (int r = 0; r < 8; ++r) {
        int m = row0 + wm * 32 + i * 16 + mh + r;
        int n = col0 + wn * 64 + j * 16 + lr;
        epi(m, n, acc[i][j][r]);
      }
}

// ---------------------------------------------------------------------------
// GEMM epilogues (fused bias / scale / scatter / activation / residual)
// ---------------------------------------------------------------------------
struct EpiQKV {           // qkv = Xw @ Wqkv + b ; scatter to [bw][3][h][NP][hd], scale q
  bf16* qkv; const float* bqkv;
  __device__ void operator()(int r, int n, float v) const {
    float val = v + bqkv[n];
    int which = n >> 9;                       // 0=q,1=k,2=v
    if (which == 0) val *= 0.17677669529663687f;   // hd^-0.5
    int h = (n >> 5) & 15;
    int d = n & 31;
    int bw = r / NP, m = r - bw * NP;
    size_t idx = ((((size_t)bw * 3 + which) * HEADS_ + h) * NP + m) * HD + d;
    qkv[idx] = (bf16)val;
  }
};

struct EpiProj {          // x1 = shortcut + window_reverse(unshift(O @ Wproj + b))
  float* out; const float* x; const float* bias;
  __device__ void operator()(int r, int n, float v) const {
    int bw = r >> 6, m = r & 63;
    int b  = bw >> 6, w = bw & 63;
    int hs = (w >> 3) * 8 + (m >> 3);
    int ws = (w & 7)  * 8 + (m & 7);
    int hh = (hs + SHIFT_) & 63;              // reverse cyclic shift
    int wc = (ws + SHIFT_) & 63;
    size_t idx = (((size_t)b * 4096) + hh * 64 + wc) * DIMC + n;
    out[idx] = x[idx] + v + bias[n];
  }
};

struct EpiGelu {          // h = gelu_exact(Xln @ W1 + b1)  -> bf16
  bf16* hbuf; const float* b1;
  __device__ void operator()(int r, int n, float v) const {
    float xv = v + b1[n];
    float g  = 0.5f * xv * (1.0f + erff(xv * 0.70710678118654752f));
    hbuf[(size_t)r * HIDDEN_ + n] = (bf16)g;
  }
};

struct EpiOutAdd {        // out += h @ W2 + b2
  float* out; const float* b2;
  __device__ void operator()(int r, int n, float v) const {
    size_t idx = (size_t)r * DIMC + n;
    out[idx] += v + b2[n];
  }
};

// ---------------------------------------------------------------------------
// Precomputed additive attention tables (branch-free hot loop, L2-resident):
//   Btab[h][64][96]: rel-pos bias; col 64 (time token) = 0; cols 65..95 = -inf
//   Mtab[w][64][96]: shifted-window mask (0 / -100); cols >= 64 = 0
// ---------------------------------------------------------------------------
__global__ __launch_bounds__(256) void build_bias_kernel(
    const float* __restrict__ relt, float* __restrict__ Btab)
{
  int i = blockIdx.x * 256 + threadIdx.x;
  if (i >= HEADS_ * 64 * 96) return;
  int n = i % 96, m = (i / 96) & 63, h = i / (96 * 64);
  float v;
  if (n >= NVALID)      v = -3.0e38f;
  else if (n == 64)     v = 0.0f;
  else {
    int dm = (m >> 3) - (n >> 3) + 7;
    int dn = (m & 7)  - (n & 7)  + 7;
    v = relt[(dm * 15 + dn) * HEADS_ + h];
  }
  Btab[i] = v;
}

__global__ __launch_bounds__(256) void build_mask_kernel(float* __restrict__ Mtab)
{
  int i = blockIdx.x * 256 + threadIdx.x;
  if (i >= 64 * 64 * 96) return;
  int n = i % 96, m = (i / 96) & 63, w = i / (96 * 64);
  float v = 0.0f;
  if (n < 64) {
    int wi = (w >> 3) * 8, wj = (w & 7) * 8;
    int rhm = wi + (m >> 3), rwm = wj + (m & 7);
    int rhn = wi + (n >> 3), rwn = wj + (n & 7);
    int cm = (rhm < 56 ? 0 : (rhm < 60 ? 1 : 2)) * 3
           + (rwm < 56 ? 0 : (rwm < 60 ? 1 : 2));
    int cn = (rhn < 56 ? 0 : (rhn < 60 ? 1 : 2)) * 3
           + (rwn < 56 ? 0 : (rwn < 60 ? 1 : 2));
    if (cm != cn) v = -100.0f;
  }
  Mtab[i] = v;
}

// ---------------------------------------------------------------------------
// Attention: one wave per (window, head). Query rows limited to the 64 real
// tokens (time-token + pad rows are dropped downstream): 4x5 wmma for Q.K^T,
// branch-free softmax with table lookups, 4x2x3 wmma for P.V.
// ---------------------------------------------------------------------------
__global__ __launch_bounds__(32) void attn_kernel(
    const bf16* __restrict__ qkv, bf16* __restrict__ obuf,
    const float* __restrict__ Btab, const float* __restrict__ Mtab)
{
  __shared__ __attribute__((aligned(64))) bf16 Pl[64 * 96];   // P (rows 0..63)
  __shared__ __attribute__((aligned(64))) bf16 Vt[HD * 96];   // V^T [d][k]

  const int lane = threadIdx.x;
  const int h  = blockIdx.x & 15;
  const int bw = blockIdx.x >> 4;
  const int w  = bw & 63;

  const bf16* Q  = qkv + (((size_t)bw * 3 + 0) * HEADS_ + h) * NP * HD;
  const bf16* Kp = qkv + (((size_t)bw * 3 + 1) * HEADS_ + h) * NP * HD;
  const bf16* V  = qkv + (((size_t)bw * 3 + 2) * HEADS_ + h) * NP * HD;
  const float* Bh = Btab + (size_t)h * 64 * 96;
  const float* Mw = Mtab + (size_t)w * 64 * 96;

  const int lr   = lane & 15;
  const int koff = (lane >> 4) * 16;
  const int mh   = (lane >> 4) * 8;

  { // zero LDS (padded tails must be 0)
    unsigned int* p = (unsigned int*)Pl;
    for (int i = lane; i < 64 * 96 / 2; i += 32) p[i] = 0u;
    unsigned int* q2 = (unsigned int*)Vt;
    for (int i = lane; i < HD * 96 / 2; i += 32) q2[i] = 0u;
  }
  // stage V^T: contiguous row reads, transposed LDS writes (rows 65..79 never
  // contribute: their P columns are exactly 0 after softmax)
  for (int k = lane; k < NP; k += 32) {
    bf16 tmp[HD];
    uint4* tp = (uint4*)tmp;
    const uint4* vr = (const uint4*)(V + (size_t)k * HD);
    tp[0] = vr[0]; tp[1] = vr[1]; tp[2] = vr[2]; tp[3] = vr[3];
    for (int d = 0; d < HD; ++d) Vt[d * 96 + k] = tmp[d];
  }

  // K fragments: lane holds key-row (lane&15)+16*tj, k-slice per lane half
  v16bf kf[5];
  for (int t = 0; t < 5; ++t)
    kf[t] = *(const v16bf*)(Kp + (size_t)(t * 16 + lr) * HD + koff);

  for (int ti = 0; ti < 4; ++ti) {              // query rows 0..63 only
    v16bf qf = *(const v16bf*)(Q + (size_t)(ti * 16 + lr) * HD + koff);
    v8f s[5];
    for (int tj = 0; tj < 5; ++tj) {
      s[tj] = v8f_zero();
      s[tj] = __builtin_amdgcn_wmma_f32_16x16x32_bf16(
          false, qf, false, kf[tj], (short)0, s[tj], false, false);
    }
    // additive bias + mask via table lookup (branch-free)
    float mx[8], sm[8];
    for (int r = 0; r < 8; ++r) mx[r] = -3.0e38f;
    for (int tj = 0; tj < 5; ++tj)
      for (int r = 0; r < 8; ++r) {
        int m = ti * 16 + mh + r;               // <= 63
        int n = tj * 16 + lr;
        int t_idx = m * 96 + n;
        float sv = s[tj][r] + Bh[t_idx] + Mw[t_idx];
        s[tj][r] = sv;
        mx[r] = fmaxf(mx[r], sv);
      }
    // row-wise softmax: xor-shuffles <=8 stay inside each 16-lane half
    for (int off = 1; off < 16; off <<= 1)
      for (int r = 0; r < 8; ++r)
        mx[r] = fmaxf(mx[r], __shfl_xor(mx[r], off, 32));
    for (int r = 0; r < 8; ++r) sm[r] = 0.0f;
    for (int tj = 0; tj < 5; ++tj)
      for (int r = 0; r < 8; ++r) {
        float e = __expf(s[tj][r] - mx[r]);
        s[tj][r] = e;
        sm[r] += e;
      }
    for (int off = 1; off < 16; off <<= 1)
      for (int r = 0; r < 8; ++r)
        sm[r] += __shfl_xor(sm[r], off, 32);
    for (int r = 0; r < 8; ++r) sm[r] = 1.0f / sm[r];
    for (int tj = 0; tj < 5; ++tj)
      for (int r = 0; r < 8; ++r) {
        int m = ti * 16 + mh + r;
        int n = tj * 16 + lr;
        Pl[m * 96 + n] = (bf16)(s[tj][r] * sm[r]);
      }
  }
  __syncthreads();

  // O = P @ V  (K padded to 96 = 3 wmma K-steps); rows 0..63 -> no store guard
  for (int ti = 0; ti < 4; ++ti) {
    v8f o0 = v8f_zero(), o1 = v8f_zero();
    for (int ks = 0; ks < 3; ++ks) {
      v16bf pa = *(const v16bf*)(Pl + (ti * 16 + lr) * 96 + ks * 32 + koff);
      v16bf b0 = *(const v16bf*)(Vt + (lr)      * 96 + ks * 32 + koff);
      v16bf b1 = *(const v16bf*)(Vt + (16 + lr) * 96 + ks * 32 + koff);
      o0 = __builtin_amdgcn_wmma_f32_16x16x32_bf16(false, pa, false, b0, (short)0, o0, false, false);
      o1 = __builtin_amdgcn_wmma_f32_16x16x32_bf16(false, pa, false, b1, (short)0, o1, false, false);
    }
    for (int r = 0; r < 8; ++r) {
      int m = ti * 16 + mh + r;
      size_t base = ((size_t)bw * 64 + m) * DIMC + h * HD;
      obuf[base + lr]      = (bf16)o0[r];
      obuf[base + 16 + lr] = (bf16)o1[r];
    }
  }
}

// ---------------------------------------------------------------------------
// Small prep kernels
// ---------------------------------------------------------------------------
__global__ __launch_bounds__(256) void time_token_kernel(
    const float* __restrict__ emb, const float* __restrict__ Wemb,
    const float* __restrict__ bemb, float* __restrict__ tbuf)
{
  const int b = blockIdx.x;
  __shared__ float sl[DIMC];
  for (int k = threadIdx.x; k < DIMC; k += 256) {
    float e = emb[b * DIMC + k];
    sl[k] = e / (1.0f + __expf(-e));        // silu
  }
  __syncthreads();
  for (int c = threadIdx.x; c < DIMC; c += 256) {
    float acc = bemb[c];
    for (int k = 0; k < DIMC; ++k) acc = fmaf(sl[k], Wemb[k * DIMC + c], acc);
    tbuf[b * DIMC + c] = acc;
  }
}

__global__ __launch_bounds__(256) void transpose_to_bf16(
    const float* __restrict__ W, bf16* __restrict__ Wt, int K, int N)
{
  int i = blockIdx.x * 256 + threadIdx.x;
  if (i >= K * N) return;
  int k = i / N, n = i - k * N;
  Wt[(size_t)n * K + k] = (bf16)W[i];
}

__device__ __forceinline__ float block512_sum(float v, float* red) {
  const int tid = threadIdx.x;
  for (int off = 16; off > 0; off >>= 1) v += __shfl_xor(v, off, 32);
  if ((tid & 31) == 0) red[tid >> 5] = v;
  __syncthreads();
  float s = 0.0f;
  for (int i = 0; i < 8; ++i) s += red[i];
  __syncthreads();
  return s;
}

// LN1 + cyclic shift + window partition + time-token append + zero pad -> bf16
__global__ __launch_bounds__(256) void prep_xw_kernel(
    const float* __restrict__ x, const float* __restrict__ g,
    const float* __restrict__ be, const float* __restrict__ tbuf,
    bf16* __restrict__ Xw)
{
  const int r  = blockIdx.x;                 // 0 .. B*NW*NP-1
  const int bw = r / NP, m = r - bw * NP;
  const int b  = bw >> 6, w = bw & 63;
  bf16* dst = Xw + (size_t)r * DIMC;
  const int tid = threadIdx.x;
  if (m >= NVALID) {                          // zero padding rows
    for (int c = tid; c < DIMC; c += 256) dst[c] = (bf16)0.0f;
    return;
  }
  if (m == 64) {                              // time-embedding token
    const float* tp = tbuf + b * DIMC;
    for (int c = tid; c < DIMC; c += 256) dst[c] = (bf16)tp[c];
    return;
  }
  const int hs = (w >> 3) * 8 + (m >> 3);
  const int ws = (w & 7)  * 8 + (m & 7);
  const int hh = (hs + SHIFT_) & 63;          // roll(xn, -SHIFT): xs[h]=xn[h+4]
  const int wc = (ws + SHIFT_) & 63;
  const float* src = x + ((size_t)b * 4096 + hh * 64 + wc) * DIMC;
  __shared__ float red[8];
  float v0 = src[tid], v1 = src[tid + 256];
  float mean = block512_sum(v0 + v1, red) * (1.0f / DIMC);
  float d0 = v0 - mean, d1 = v1 - mean;
  float var = block512_sum(d0 * d0 + d1 * d1, red) * (1.0f / DIMC);
  float rstd = rsqrtf(var + 1e-5f);
  dst[tid]       = (bf16)(d0 * rstd * g[tid]       + be[tid]);
  dst[tid + 256] = (bf16)(d1 * rstd * g[tid + 256] + be[tid + 256]);
}

__global__ __launch_bounds__(256) void ln2_kernel(
    const float* __restrict__ x1, const float* __restrict__ g,
    const float* __restrict__ be, bf16* __restrict__ Xln)
{
  const int r = blockIdx.x;                   // 0 .. 65535
  const float* src = x1 + (size_t)r * DIMC;
  bf16* dst = Xln + (size_t)r * DIMC;
  const int tid = threadIdx.x;
  __shared__ float red[8];
  float v0 = src[tid], v1 = src[tid + 256];
  float mean = block512_sum(v0 + v1, red) * (1.0f / DIMC);
  float d0 = v0 - mean, d1 = v1 - mean;
  float var = block512_sum(d0 * d0 + d1 * d1, red) * (1.0f / DIMC);
  float rstd = rsqrtf(var + 1e-5f);
  dst[tid]       = (bf16)(d0 * rstd * g[tid]       + be[tid]);
  dst[tid + 256] = (bf16)(d1 * rstd * g[tid + 256] + be[tid + 256]);
}

// ---------------------------------------------------------------------------
// Orchestration
// ---------------------------------------------------------------------------
extern "C" void kernel_launch(void* const* d_in, const int* in_sizes, int n_in,
                              void* d_out, int out_size, void* d_ws, size_t ws_size,
                              hipStream_t stream)
{
  (void)in_sizes; (void)n_in; (void)out_size; (void)ws_size;
  const float* x      = (const float*)d_in[0];
  const float* emb    = (const float*)d_in[1];
  const float* gamma1 = (const float*)d_in[2];
  const float* beta1  = (const float*)d_in[3];
  const float* W_emb  = (const float*)d_in[4];
  const float* b_emb  = (const float*)d_in[5];
  const float* W_qkv  = (const float*)d_in[6];
  const float* b_qkv  = (const float*)d_in[7];
  const float* relt   = (const float*)d_in[8];
  const float* W_proj = (const float*)d_in[9];
  const float* b_proj = (const float*)d_in[10];
  const float* gamma2 = (const float*)d_in[11];
  const float* beta2  = (const float*)d_in[12];
  const float* W1     = (const float*)d_in[13];
  const float* b1     = (const float*)d_in[14];
  const float* W2     = (const float*)d_in[15];
  const float* b2     = (const float*)d_in[16];
  float* out = (float*)d_out;

  char* ws = (char*)d_ws;
  size_t off = 0;
  auto alloc = [&](size_t bytes) -> char* {
    char* p = ws + off;
    off += (bytes + 255) & ~(size_t)255;
    return p;
  };
  float* tbuf  = (float*)alloc((size_t)16 * 512 * sizeof(float));
  bf16* wqkvT  = (bf16*)alloc((size_t)1536 * 512 * 2);
  bf16* wprojT = (bf16*)alloc((size_t)512 * 512 * 2);
  bf16* w1T    = (bf16*)alloc((size_t)2048 * 512 * 2);
  bf16* w2T    = (bf16*)alloc((size_t)512 * 2048 * 2);
  float* Btab  = (float*)alloc((size_t)HEADS_ * 64 * 96 * sizeof(float));
  float* Mtab  = (float*)alloc((size_t)64 * 64 * 96 * sizeof(float));

  const size_t XW_BYTES  = (size_t)81920 * 512 * 2;                 // Xw  (84 MB)
  const size_t QKV_BYTES = (size_t)1024 * 3 * HEADS_ * NP * HD * 2; // qkv (252 MB)
  const size_t H_BYTES   = (size_t)65536 * HIDDEN_ * 2;             // Hbuf (268 MB)
  size_t regA = XW_BYTES + QKV_BYTES; if (H_BYTES > regA) regA = H_BYTES;
  char* regionA = alloc(regA);                 // Xw+qkv, later reused as Hbuf
  char* regionB = alloc((size_t)65536 * 512 * 2);  // Obuf, later reused as Xln

  bf16* Xw   = (bf16*)regionA;
  bf16* qkvb = (bf16*)(regionA + XW_BYTES);
  bf16* Hbuf = (bf16*)regionA;                 // alias: valid after attn+proj done
  bf16* Obuf = (bf16*)regionB;
  bf16* Xln  = (bf16*)regionB;                 // alias: valid after proj done

  // prep
  time_token_kernel<<<16, 256, 0, stream>>>(emb, W_emb, b_emb, tbuf);
  transpose_to_bf16<<<(512 * 1536 + 255) / 256, 256, 0, stream>>>(W_qkv, wqkvT, 512, 1536);
  transpose_to_bf16<<<(512 * 512 + 255) / 256, 256, 0, stream>>>(W_proj, wprojT, 512, 512);
  transpose_to_bf16<<<(512 * 2048 + 255) / 256, 256, 0, stream>>>(W1, w1T, 512, 2048);
  transpose_to_bf16<<<(2048 * 512 + 255) / 256, 256, 0, stream>>>(W2, w2T, 2048, 512);
  build_bias_kernel<<<(HEADS_ * 64 * 96 + 255) / 256, 256, 0, stream>>>(relt, Btab);
  build_mask_kernel<<<(64 * 64 * 96 + 255) / 256, 256, 0, stream>>>(Mtab);
  prep_xw_kernel<<<81920, 256, 0, stream>>>(x, gamma1, beta1, tbuf, Xw);

  // QKV GEMM: [81920,512] x [512,1536]
  EpiQKV e1{qkvb, b_qkv};
  gemm_bf16_wmma<<<dim3(1536 / 128, 81920 / 128), 256, 0, stream>>>(
      Xw, wqkvT, 81920, 1536, 512, e1);

  // windowed attention: one wave per (window, head)
  attn_kernel<<<16 * 64 * HEADS_, 32, 0, stream>>>(qkvb, Obuf, Btab, Mtab);

  // proj GEMM + residual + window reverse + unshift -> out (= x1)
  EpiProj e2{out, x, b_proj};
  gemm_bf16_wmma<<<dim3(512 / 128, 65536 / 128), 256, 0, stream>>>(
      Obuf, wprojT, 65536, 512, 512, e2);

  // LN2 -> bf16
  ln2_kernel<<<65536, 256, 0, stream>>>(out, gamma2, beta2, Xln);

  // MLP1 GEMM + exact GELU -> bf16
  EpiGelu e3{Hbuf, b1};
  gemm_bf16_wmma<<<dim3(2048 / 128, 65536 / 128), 256, 0, stream>>>(
      Xln, w1T, 65536, 2048, 512, e3);

  // MLP2 GEMM + residual add into out
  EpiOutAdd e4{out, b2};
  gemm_bf16_wmma<<<dim3(512 / 128, 65536 / 128), 256, 0, stream>>>(
      Hbuf, w2T, 65536, 512, 2048, e4);
}
